// multichannel_network_7791070675759
// MI455X (gfx1250) — matrix operand order repaired and checked by reading, not compiled, not run on hardware
//
#include <hip/hip_runtime.h>
#include <math.h>

// ---------------------------------------------------------------------------
// Types for CDNA5 WMMA
// ---------------------------------------------------------------------------
typedef __bf16 bf16_t;
typedef bf16_t v16bf __attribute__((ext_vector_type(16)));
typedef bf16_t v2bf  __attribute__((ext_vector_type(2)));
typedef float  v8f   __attribute__((ext_vector_type(8)));

#define BN_EPS 0.001f
#define GAMMA_ 0.001f

#define MODE_SIM     1
#define MODE_RELU_BN 2
#define MODE_BN      3

// LDS tile: 64 rows x 32 bf16, padded to 40 bf16 (20 dwords, 80B) per row so
// every 16B fragment chunk stays 16B-aligned and banks are spread.
#define LDS_STRIDE_U 20
#define LDS_STRIDE_H 40

// Native RNE conversions: clang lowers (float)->(__bf16) to v_cvt_bf16_f32 and
// fuses pairs into v_cvt_pk_bf16_f32 -- far cheaper than manual bit rounding.
__device__ __forceinline__ unsigned int pkbf(float lo, float hi) {
  union { v2bf v; unsigned int u; } r;
  r.v.x = (bf16_t)lo;
  r.v.y = (bf16_t)hi;
  return r.u;
}
__device__ __forceinline__ bf16_t f2bf(float f) { return (bf16_t)f; }

// ---------------------------------------------------------------------------
// Row sum-of-squares: out[row] = sum_c X[row][c]^2   (one block per row)
// ---------------------------------------------------------------------------
__global__ __launch_bounds__(256) void rownorm_kernel(const float* __restrict__ X,
                                                      int cols,
                                                      float* __restrict__ out) {
  const int row = blockIdx.x;
  const float* r = X + (size_t)row * cols;
  const float4* r4 = (const float4*)r;
  const int c4 = cols >> 2;
  float s = 0.f;
  for (int c = threadIdx.x; c < c4; c += blockDim.x) {
    const float4 v = r4[c];
    s += v.x * v.x + v.y * v.y + v.z * v.z + v.w * v.w;
  }
  for (int c = (c4 << 2) + threadIdx.x; c < cols; c += blockDim.x) {
    const float v = r[c]; s += v * v;
  }
  __shared__ float red[256];
  red[threadIdx.x] = s;
  __syncthreads();
  for (int st = 128; st > 0; st >>= 1) {
    if ((int)threadIdx.x < st) red[threadIdx.x] += red[threadIdx.x + st];
    __syncthreads();
  }
  if (threadIdx.x == 0) out[row] = red[0];
}

// ---------------------------------------------------------------------------
// Unified bf16 WMMA GEMM: C(M,N) = A(M,K) @ B
//   BTRANS=0: B is (N,K) row-major (S matrix; C = A @ S^T), ldB = K
//             -> software-pipelined streaming path (HBM-bound on MI455X)
//   BTRANS=1: B is (K,N) row-major (weight W; C = A @ W),   ldB = N
// Fused epilogues (MODE_SIM / MODE_RELU_BN / MODE_BN), output written to
// C[m*ldC + ncolofs + n] so concat slices are free.
// Tile: BM=BN=64, BK=32, 256 threads (8 wave32), each wave owns two 16x16
// f32 accumulators fed by v_wmma_f32_16x16x32_bf16.
// ---------------------------------------------------------------------------
template <int BTRANS>
__global__ __launch_bounds__(256) void wmma_gemm_kernel(
    const float* __restrict__ A, const float* __restrict__ Bm,
    int M, int N, int K, int ldB,
    float* __restrict__ C, int ldC, int ncolofs, int mode,
    const float* __restrict__ bias,
    const float* __restrict__ bng, const float* __restrict__ bnb,
    const float* __restrict__ bnm, const float* __restrict__ bnv,
    const float* __restrict__ xx, const float* __restrict__ ss) {
  __shared__ unsigned int AsU[64 * LDS_STRIDE_U];  // [m][k] as bf16 pairs
  __shared__ unsigned int BsU[64 * LDS_STRIDE_U];  // [n][k] as bf16 pairs

  const int tid  = threadIdx.x;
  const int wave = tid >> 5;
  const int lane = tid & 31;
  const int m0 = blockIdx.y * 64;
  const int n0 = blockIdx.x * 64;

  // Wave -> (m-subtile, 2 consecutive n-subtiles)
  const int msub = wave & 3;
  const int nsb  = (wave >> 2) * 2;

  v8f c0 = {}; v8f c1 = {};

  // LDS fill mapping: 4 threads per row, 8 contiguous k each
  const int frow = tid >> 2;
  const int fk   = (tid & 3) * 8;
  const int gm   = (m0 + frow < M) ? (m0 + frow) : (M - 1);
  const int gnB  = (n0 + frow < N) ? (n0 + frow) : (N - 1);  // clamp; OOB cols discarded in epilogue

  // Fragment lane mapping (contiguous k order k = 16*hi + s, same for A and B)
  const int mrow = lane & 15;   // A: row M, B: col N
  const int hi   = lane >> 4;
  const int akb  = hi * 16;     // A: lanes 16-31 start at K offset 16
  const int bkb  = hi * 16;     // B: lanes 16-31 start at K offset 16
  const int abase  = (msub * 16 + mrow) * LDS_STRIDE_U + (akb >> 1);
  const int bbase0 = ((nsb + 0) * 16 + mrow) * LDS_STRIDE_U + (bkb >> 1);
  const int bbase1 = ((nsb + 1) * 16 + mrow) * LDS_STRIDE_U + (bkb >> 1);

#define COMPUTE_TILE()                                                          \
  do {                                                                          \
    union { v16bf v; uint4 q[2]; } af, bf0, bf1;                                \
    af.q[0]  = *(const uint4*)&AsU[abase];                                      \
    af.q[1]  = *(const uint4*)&AsU[abase + 4];                                  \
    bf0.q[0] = *(const uint4*)&BsU[bbase0];                                     \
    bf0.q[1] = *(const uint4*)&BsU[bbase0 + 4];                                 \
    bf1.q[0] = *(const uint4*)&BsU[bbase1];                                     \
    bf1.q[1] = *(const uint4*)&BsU[bbase1 + 4];                                 \
    c0 = __builtin_amdgcn_wmma_f32_16x16x32_bf16(false, af.v, false, bf0.v,     \
                                                 (short)0, c0, false, false);   \
    c1 = __builtin_amdgcn_wmma_f32_16x16x32_bf16(false, af.v, false, bf1.v,     \
                                                 (short)0, c1, false, false);   \
  } while (0)

  const int kfull = K & ~31;

  if (BTRANS == 0) {
    // ---- streaming path: reg-staged software pipeline, branch-free fill ----
    float4 ra0, ra1, rb0, rb1;
    if (kfull > 0) {
      {
        const float* pa = A  + (size_t)gm  * K   + fk;
        const float* pb = Bm + (size_t)gnB * ldB + fk;
        ra0 = *(const float4*)pa; ra1 = *(const float4*)(pa + 4);
        rb0 = *(const float4*)pb; rb1 = *(const float4*)(pb + 4);
      }
      for (int k0 = 0; k0 < kfull; k0 += 32) {
        // store staged regs -> LDS (one b128 store per matrix per thread)
        {
          uint4 ua, ub;
          ua.x = pkbf(ra0.x, ra0.y); ua.y = pkbf(ra0.z, ra0.w);
          ua.z = pkbf(ra1.x, ra1.y); ua.w = pkbf(ra1.z, ra1.w);
          ub.x = pkbf(rb0.x, rb0.y); ub.y = pkbf(rb0.z, rb0.w);
          ub.z = pkbf(rb1.x, rb1.y); ub.w = pkbf(rb1.z, rb1.w);
          *(uint4*)&AsU[frow * LDS_STRIDE_U + (fk >> 1)] = ua;
          *(uint4*)&BsU[frow * LDS_STRIDE_U + (fk >> 1)] = ub;
        }
        __syncthreads();
        // issue next tile's global loads; they fly while WMMAs run
        if (k0 + 32 < kfull) {
          const float* pa = A  + (size_t)gm  * K   + (k0 + 32 + fk);
          const float* pb = Bm + (size_t)gnB * ldB + (k0 + 32 + fk);
          ra0 = *(const float4*)pa; ra1 = *(const float4*)(pa + 4);
          rb0 = *(const float4*)pb; rb1 = *(const float4*)(pb + 4);
        }
        COMPUTE_TILE();
        __syncthreads();
      }
    }
    if (kfull < K) {  // K-remainder tail: clamped loads + select-zero (no exec branching)
      #pragma unroll
      for (int i = 0; i < 8; i += 2) {
        const int gk0 = kfull + fk + i, gk1 = gk0 + 1;
        const int ck0 = gk0 < K ? gk0 : K - 1, ck1 = gk1 < K ? gk1 : K - 1;
        float a0v = A[(size_t)gm * K + ck0], a1v = A[(size_t)gm * K + ck1];
        float b0v = Bm[(size_t)gnB * ldB + ck0], b1v = Bm[(size_t)gnB * ldB + ck1];
        a0v = gk0 < K ? a0v : 0.f; a1v = gk1 < K ? a1v : 0.f;
        b0v = gk0 < K ? b0v : 0.f; b1v = gk1 < K ? b1v : 0.f;
        AsU[frow * LDS_STRIDE_U + ((fk + i) >> 1)] = pkbf(a0v, a1v);
        BsU[frow * LDS_STRIDE_U + ((fk + i) >> 1)] = pkbf(b0v, b1v);
      }
      __syncthreads();
      COMPUTE_TILE();
      __syncthreads();
    }
  } else {
    // ---- weight path: W is (K,N); coalesced row loads, transpose via LDS ----
    const int kr   = tid >> 3;        // 0..31 (k within tile)
    const int nseg = (tid & 7) * 8;   // 8 consecutive n (N here is 512/128, full tiles)
    bf16_t* Bh = (bf16_t*)BsU;
    for (int k0 = 0; k0 < K; k0 += 32) {
      // A fill (same layout as streaming path)
      if (k0 + 32 <= K) {
        const float* pa = A + (size_t)gm * K + k0 + fk;
        const float4 a0 = *(const float4*)pa;
        const float4 a1 = *(const float4*)(pa + 4);
        uint4 ua;
        ua.x = pkbf(a0.x, a0.y); ua.y = pkbf(a0.z, a0.w);
        ua.z = pkbf(a1.x, a1.y); ua.w = pkbf(a1.z, a1.w);
        *(uint4*)&AsU[frow * LDS_STRIDE_U + (fk >> 1)] = ua;
      } else {
        #pragma unroll
        for (int i = 0; i < 8; i += 2) {
          const int gk0 = k0 + fk + i, gk1 = gk0 + 1;
          const int ck0 = gk0 < K ? gk0 : K - 1, ck1 = gk1 < K ? gk1 : K - 1;
          float a0v = A[(size_t)gm * K + ck0], a1v = A[(size_t)gm * K + ck1];
          a0v = gk0 < K ? a0v : 0.f; a1v = gk1 < K ? a1v : 0.f;
          AsU[frow * LDS_STRIDE_U + ((fk + i) >> 1)] = pkbf(a0v, a1v);
        }
      }
      // B fill: read W row (contiguous n), scatter-transpose into [n][k]
      {
        const int gk = k0 + kr;
        const int ck = gk < K ? gk : K - 1;
        const float* pb = Bm + (size_t)ck * ldB + n0 + nseg;
        float4 b0 = *(const float4*)pb;
        float4 b1 = *(const float4*)(pb + 4);
        if (gk >= K) { b0 = make_float4(0.f, 0.f, 0.f, 0.f); b1 = b0; }
        Bh[(nseg + 0) * LDS_STRIDE_H + kr] = f2bf(b0.x);
        Bh[(nseg + 1) * LDS_STRIDE_H + kr] = f2bf(b0.y);
        Bh[(nseg + 2) * LDS_STRIDE_H + kr] = f2bf(b0.z);
        Bh[(nseg + 3) * LDS_STRIDE_H + kr] = f2bf(b0.w);
        Bh[(nseg + 4) * LDS_STRIDE_H + kr] = f2bf(b1.x);
        Bh[(nseg + 5) * LDS_STRIDE_H + kr] = f2bf(b1.y);
        Bh[(nseg + 6) * LDS_STRIDE_H + kr] = f2bf(b1.z);
        Bh[(nseg + 7) * LDS_STRIDE_H + kr] = f2bf(b1.w);
      }
      __syncthreads();
      COMPUTE_TILE();
      __syncthreads();
    }
  }
#undef COMPUTE_TILE

  // Epilogue. C/D layout: VGPR i, lane l -> row = i + 8*(l>=16), col = l&15
  #pragma unroll
  for (int s = 0; s < 2; ++s) {
    const v8f c = s ? c1 : c0;
    const int n = n0 + (nsb + s) * 16 + mrow;
    if (n >= N) continue;
    #pragma unroll
    for (int i = 0; i < 8; ++i) {
      const int m = m0 + msub * 16 + i + hi * 8;
      if (m >= M) continue;
      float v = c[i];
      if (mode == MODE_SIM) {
        const float d2 = xx[m] + ss[n] - 2.f * v;
        v = __expf(-GAMMA_ * fmaxf(d2, 0.f));
      } else {
        v += bias[n];
        if (mode == MODE_RELU_BN) v = fmaxf(v, 0.f);
        v = (v - bnm[n]) * rsqrtf(bnv[n] + BN_EPS) * bng[n] + bnb[n];
      }
      C[(size_t)m * ldC + (ncolofs + n)] = v;
    }
  }
}

// ---------------------------------------------------------------------------
// Fused conv1d(VALID,win=8) + bias + relu + BN + maxpool3(stride3,VALID).
// Layouts: in (B,Lin,Cin), w (8,Cin,Cout), out (B,Lp,Cout).
// ---------------------------------------------------------------------------
__global__ __launch_bounds__(256) void conv_relu_bn_pool_kernel(
    const float* __restrict__ in, const float* __restrict__ w,
    const float* __restrict__ bias,
    const float* __restrict__ bng, const float* __restrict__ bnb,
    const float* __restrict__ bnm, const float* __restrict__ bnv,
    float* __restrict__ out, int B, int Lin, int Cin, int Cout, int Lp) {
  const int idx = blockIdx.x * blockDim.x + threadIdx.x;
  const int total = B * Lp * Cout;
  if (idx >= total) return;
  const int co = idx % Cout;
  int t = idx / Cout;
  const int lp = t % Lp;
  const int b  = t / Lp;
  const float scale = bng[co] * rsqrtf(bnv[co] + BN_EPS);
  const float shift = bnb[co] - bnm[co] * scale;
  float mx = -3.402823466e+38f;
  for (int p = 0; p < 3; ++p) {
    const int pos = lp * 3 + p;
    float acc = bias[co];
    for (int tw = 0; tw < 8; ++tw)
      for (int ci = 0; ci < Cin; ++ci)
        acc += in[((size_t)b * Lin + (pos + tw)) * Cin + ci] * w[(tw * Cin + ci) * Cout + co];
    const float v = fmaxf(acc, 0.f) * scale + shift;
    mx = fmaxf(mx, v);
  }
  out[idx] = mx;
}

// ---------------------------------------------------------------------------
// Final layer: out[b] = sigmoid(dot(h[b,0:128], w) + bias). One block / batch.
// ---------------------------------------------------------------------------
__global__ __launch_bounds__(128) void final_dot_kernel(const float* __restrict__ h,
                                                        const float* __restrict__ w,
                                                        const float* __restrict__ b,
                                                        float* __restrict__ out) {
  const int bi = blockIdx.x;
  const int t = threadIdx.x;
  __shared__ float red[128];
  red[t] = h[(size_t)bi * 128 + t] * w[t];
  __syncthreads();
  for (int s = 64; s > 0; s >>= 1) {
    if (t < s) red[t] += red[t + s];
    __syncthreads();
  }
  if (t == 0) out[bi] = 1.f / (1.f + __expf(-(red[0] + b[0])));
}

// ---------------------------------------------------------------------------
// Host launcher
// ---------------------------------------------------------------------------
extern "C" void kernel_launch(void* const* d_in, const int* in_sizes, int n_in,
                              void* d_out, int out_size, void* d_ws, size_t ws_size,
                              hipStream_t stream) {
  (void)in_sizes; (void)n_in; (void)out_size; (void)ws_size;
  auto F = [&](int i) { return (const float*)d_in[i]; };

  // --- inputs (setup_inputs dict order) ---
  const float* fingerprint = F(0);   // (256, 881)
  const float* rdkit = F(1);         // (256, 200)
  const float* cnv   = F(2);         // (256, 20000)
  const float* ge    = F(3);         // (256, 17000)
  const float* mut   = F(4);         // (256, 3000)
  /* meth = F(5): branch result discarded by the reference; skipped */
  const float* S_cnv = F(6);         // (1000, 20000)
  const float* S_ge  = F(7);         // (1000, 17000)
  const float* S_mut = F(8);         // (1000, 3000)
  /* S_meth = F(9) unused */

  // --- params: pytree (sorted keys) leaves, base index 10 ---
  // bn dicts: beta,gamma,mean,var ; dense/conv dicts: b,w
  // bn1=10 bn2=14 bn3=18 bn4=22 bn5=26
  // cnv1.b=30 cnv1.w=31 cnv2.b=32 cnv2.w=33 cnv_bn1=34 cnv_bn2=38
  // conv1.b=42 conv1.w=43 conv2.b=44 conv2.w=45 conv3.b=46 conv3.w=47
  // fc1.b=48 fc1.w=49 fc2.b=50 fc2.w=51
  // ge1.b=52 ge1.w=53 ge2.b=54 ge2.w=55 ge_bn1=56 ge_bn2=60
  // int1.b=64 int1.w=65 int2.b=66 int2.w=67 int3.b=68 int3.w=69
  // int_bn1=70 int_bn2=74
  // meth* = 78..89 (unused)
  // mut1.b=90 mut1.w=91 mut2.b=92 mut2.w=93 mut_bn1=94 mut_bn2=98
  // rdkit1.b=102 rdkit1.w=103 rdkit2.b=104 rdkit2.w=105 rdkit_bn1=106 rdkit_bn2=110

  float* ws = (float*)d_ws;
  size_t off = 0;
  auto alloc = [&](size_t n) { size_t o = off; off += n; return o; };
  const size_t oG_cnv = alloc(256 * 1000);
  const size_t oG_ge  = alloc(256 * 1000);
  const size_t oG_mut = alloc(256 * 1000);
  const size_t oxx_cnv = alloc(256), oss_cnv = alloc(1000);
  const size_t oxx_ge  = alloc(256), oss_ge  = alloc(1000);
  const size_t oxx_mut = alloc(256), oss_mut = alloc(1000);
  const size_t ohc = alloc(256 * 512);
  const size_t ohg = alloc(256 * 512);
  const size_t ohm = alloc(256 * 512);
  const size_t ohr = alloc(256 * 512);
  const size_t oc1 = alloc(256 * 291 * 4);
  const size_t oc2 = alloc(256 * 94 * 8);
  const size_t oc3 = alloc(256 * 29 * 16);   // flatten -> (256, 464)
  const size_t ofc1 = alloc(256 * 512);
  const size_t ofeat = alloc(256 * 640);
  const size_t oi1 = alloc(256 * 512);
  const size_t oi2 = alloc(256 * 128);

  const dim3 blk(256);
  const dim3 gSim(16, 4);   // N=1000 -> 16 tiles, M=256 -> 4
  const dim3 g512(8, 4);    // N=512
  const dim3 g128(2, 4);    // N=128
  const float* nil = nullptr;

  // 1) row norms (also warms L2 for the similarity GEMMs that follow)
  rownorm_kernel<<<256, blk, 0, stream>>>(cnv, 20000, ws + oxx_cnv);
  rownorm_kernel<<<1000, blk, 0, stream>>>(S_cnv, 20000, ws + oss_cnv);
  rownorm_kernel<<<256, blk, 0, stream>>>(ge, 17000, ws + oxx_ge);
  rownorm_kernel<<<1000, blk, 0, stream>>>(S_ge, 17000, ws + oss_ge);
  rownorm_kernel<<<256, blk, 0, stream>>>(mut, 3000, ws + oxx_mut);
  rownorm_kernel<<<1000, blk, 0, stream>>>(S_mut, 3000, ws + oss_mut);

  // 2) similarity GEMMs with fused exp(-gamma*max(d2,0)) epilogue
  wmma_gemm_kernel<0><<<gSim, blk, 0, stream>>>(cnv, S_cnv, 256, 1000, 20000, 20000,
      ws + oG_cnv, 1000, 0, MODE_SIM, nil, nil, nil, nil, nil, ws + oxx_cnv, ws + oss_cnv);
  wmma_gemm_kernel<0><<<gSim, blk, 0, stream>>>(ge, S_ge, 256, 1000, 17000, 17000,
      ws + oG_ge, 1000, 0, MODE_SIM, nil, nil, nil, nil, nil, ws + oxx_ge, ws + oss_ge);
  wmma_gemm_kernel<0><<<gSim, blk, 0, stream>>>(mut, S_mut, 256, 1000, 3000, 3000,
      ws + oG_mut, 1000, 0, MODE_SIM, nil, nil, nil, nil, nil, ws + oxx_mut, ws + oss_mut);

  // 3) omics dense stacks (relu+bn), second layer writes into feat slice
  wmma_gemm_kernel<1><<<g512, blk, 0, stream>>>(ws + oG_cnv, F(31), 256, 512, 1000, 512,
      ws + ohc, 512, 0, MODE_RELU_BN, F(30), F(35), F(34), F(36), F(37), nil, nil);
  wmma_gemm_kernel<1><<<g128, blk, 0, stream>>>(ws + ohc, F(33), 256, 128, 512, 128,
      ws + ofeat, 640, 256, MODE_RELU_BN, F(32), F(39), F(38), F(40), F(41), nil, nil);
  wmma_gemm_kernel<1><<<g512, blk, 0, stream>>>(ws + oG_ge, F(53), 256, 512, 1000, 512,
      ws + ohg, 512, 0, MODE_RELU_BN, F(52), F(57), F(56), F(58), F(59), nil, nil);
  wmma_gemm_kernel<1><<<g128, blk, 0, stream>>>(ws + ohg, F(55), 256, 128, 512, 128,
      ws + ofeat, 640, 384, MODE_RELU_BN, F(54), F(61), F(60), F(62), F(63), nil, nil);
  wmma_gemm_kernel<1><<<g512, blk, 0, stream>>>(ws + oG_mut, F(91), 256, 512, 1000, 512,
      ws + ohm, 512, 0, MODE_RELU_BN, F(90), F(95), F(94), F(96), F(97), nil, nil);
  wmma_gemm_kernel<1><<<g128, blk, 0, stream>>>(ws + ohm, F(93), 256, 128, 512, 128,
      ws + ofeat, 640, 512, MODE_RELU_BN, F(92), F(99), F(98), F(100), F(101), nil, nil);

  // 4) rdkit stack
  wmma_gemm_kernel<1><<<g512, blk, 0, stream>>>(rdkit, F(103), 256, 512, 200, 512,
      ws + ohr, 512, 0, MODE_RELU_BN, F(102), F(107), F(106), F(108), F(109), nil, nil);
  wmma_gemm_kernel<1><<<g128, blk, 0, stream>>>(ws + ohr, F(105), 256, 128, 512, 128,
      ws + ofeat, 640, 128, MODE_RELU_BN, F(104), F(111), F(110), F(112), F(113), nil, nil);

  // 5) drug CNN branch: 881 ->(conv8,pool3)-> 291x4 -> 94x8 -> 29x16 = 464
  conv_relu_bn_pool_kernel<<<(256 * 291 * 4 + 255) / 256, blk, 0, stream>>>(
      fingerprint, F(43), F(42), F(11), F(10), F(12), F(13),
      ws + oc1, 256, 881, 1, 4, 291);
  conv_relu_bn_pool_kernel<<<(256 * 94 * 8 + 255) / 256, blk, 0, stream>>>(
      ws + oc1, F(45), F(44), F(15), F(14), F(16), F(17),
      ws + oc2, 256, 291, 4, 8, 94);
  conv_relu_bn_pool_kernel<<<(256 * 29 * 16 + 255) / 256, blk, 0, stream>>>(
      ws + oc2, F(47), F(46), F(19), F(18), F(20), F(21),
      ws + oc3, 256, 94, 8, 16, 29);
  wmma_gemm_kernel<1><<<g512, blk, 0, stream>>>(ws + oc3, F(49), 256, 512, 464, 512,
      ws + ofc1, 512, 0, MODE_BN, F(48), F(23), F(22), F(24), F(25), nil, nil);
  wmma_gemm_kernel<1><<<g128, blk, 0, stream>>>(ws + ofc1, F(51), 256, 128, 512, 128,
      ws + ofeat, 640, 0, MODE_BN, F(50), F(27), F(26), F(28), F(29), nil, nil);

  // 6) integration head
  wmma_gemm_kernel<1><<<g512, blk, 0, stream>>>(ws + ofeat, F(65), 256, 512, 640, 512,
      ws + oi1, 512, 0, MODE_RELU_BN, F(64), F(71), F(70), F(72), F(73), nil, nil);
  wmma_gemm_kernel<1><<<g128, blk, 0, stream>>>(ws + oi1, F(67), 256, 128, 512, 128,
      ws + oi2, 128, 0, MODE_RELU_BN, F(66), F(75), F(74), F(76), F(77), nil, nil);
  final_dot_kernel<<<256, 128, 0, stream>>>(ws + oi2, F(69), F(68), (float*)d_out);
}
